// ParametricKernelLayer_62268435857633
// MI455X (gfx1250) — matrix operand order repaired and checked by reading, not compiled, not run on hardware
//
#include <hip/hip_runtime.h>
#include <hip/hip_bf16.h>
#include <math.h>

// ---------------------------------------------------------------------------
// ParametricKernelLayer for MI455X (gfx1250, wave32, WMMA)
//
// Shapes: B=16, F=4, K=16, Y=16, X=16, P=16, T=8   (PT = 128, YX = 256)
//
// Kernel 1 (1 block): batch means, 1D separable kernels, normalization
//                     integrals (all exact f32, separability exploited).
// Kernel 2 (64 blocks = one per (b,f), 128 threads = 4 waves):
//   Stage A: H[yx,k] = sum_pt field[b,f,yx,pt] * C[b,f,k,pt]
//            as M=256, N=16, inner=128 GEMM via v_wmma_f32_16x16x32_f16
//            (4 M-tiles per wave, 4 K-steps per tile, f32 accumulate).
//   Stage B: feats[b,f,k] = sum_yx H[yx,k]*klat[k,y]*klon[k,x]*darea[b,yx]
//            done in f32 VALU on the accumulator fragments + LDS reduction,
//            then scaled by 1/(integral+eps).
// ---------------------------------------------------------------------------

typedef __attribute__((ext_vector_type(16))) _Float16 v16h;
typedef __attribute__((ext_vector_type(8)))  float    v8f;

#define NB 16
#define NF 4
#define NK 16
#define NY 16
#define NX 16
#define NP 16
#define NT 8
#define NPT (NP * NT)   // 128
#define NYX (NY * NX)   // 256

// workspace layout (floats)
#define WS_KLAT 0                       // [F*K*16]
#define WS_KLON (WS_KLAT + NF*NK*NY)    // [F*K*16]
#define WS_KLEV (WS_KLON + NF*NK*NX)    // [F*K*16]
#define WS_KTIME (WS_KLEV + NF*NK*NP)   // [F*K*8]
#define WS_INV  (WS_KTIME + NF*NK*NT)   // [F*K]

// ---------------------------------------------------------------------------
// Kernel 1: means + 1D kernels + normalization integrals (single block, 256 thr)
// ---------------------------------------------------------------------------
__global__ __launch_bounds__(256) void pkl_prep_kernel(
    const float* __restrict__ darea,   // (B,Y,X)
    const float* __restrict__ dlev,    // (B,P)
    const float* __restrict__ dtime,   // (B,T)
    const float* __restrict__ lat_mean, const float* __restrict__ lat_logstd,
    const float* __restrict__ lon_mean, const float* __restrict__ lon_logstd,
    const float* __restrict__ lev_mean, const float* __restrict__ lev_logstd,
    const float* __restrict__ time_logtau,
    float* __restrict__ ws)
{
    __shared__ float smArea[NYX];
    __shared__ float smLev[NP];
    __shared__ float smTime[NT];

    const int tid = threadIdx.x;

    // batch means
    {
        float s = 0.f;
        #pragma unroll
        for (int b = 0; b < NB; ++b) s += darea[b * NYX + tid];
        smArea[tid] = s * (1.f / NB);
    }
    if (tid < NP) {
        float s = 0.f;
        #pragma unroll
        for (int b = 0; b < NB; ++b) s += dlev[b * NP + tid];
        smLev[tid] = s * (1.f / NB);
    }
    if (tid < NT) {
        float s = 0.f;
        #pragma unroll
        for (int b = 0; b < NB; ++b) s += dtime[b * NT + tid];
        smTime[tid] = s * (1.f / NB);
    }
    __syncthreads();

    if (tid < NF * NK) {
        const int fk = tid;   // f*16 + k ; params are flat (F,K)

        // 1D kernels (linspace(-1,1,L): coord = -1 + 2*i/(L-1))
        float klat[NY], klon[NX], klev[NP], ktime[NT];
        {
            const float m = lat_mean[fk], is = 1.f / expf(lat_logstd[fk]);
            #pragma unroll
            for (int y = 0; y < NY; ++y) {
                float c = (-1.f + (2.f / (NY - 1)) * y - m) * is;
                klat[y] = expf(-0.5f * c * c);
            }
        }
        {
            const float m = lon_mean[fk], is = 1.f / expf(lon_logstd[fk]);
            #pragma unroll
            for (int x = 0; x < NX; ++x) {
                float c = (-1.f + (2.f / (NX - 1)) * x - m) * is;
                klon[x] = expf(-0.5f * c * c);
            }
        }
        {
            const float m = lev_mean[fk], is = 1.f / expf(lev_logstd[fk]);
            #pragma unroll
            for (int p = 0; p < NP; ++p) {
                float c = (-1.f + (2.f / (NP - 1)) * p - m) * is;
                klev[p] = expf(-0.5f * c * c);
            }
        }
        {
            const float itau = 1.f / (expf(time_logtau[fk]) + 1e-4f);
            #pragma unroll
            for (int t = 0; t < NT; ++t) ktime[t] = expf(-(float)t * itau);
        }

        // integral = (klat^T . dareamean . klon) * (klev.dlevmean) * (ktime.dtimemean)
        float Sarea = 0.f;
        #pragma unroll
        for (int y = 0; y < NY; ++y) {
            float row = 0.f;
            #pragma unroll
            for (int x = 0; x < NX; ++x) row += klon[x] * smArea[y * NX + x];
            Sarea += klat[y] * row;
        }
        float Slev = 0.f;
        #pragma unroll
        for (int p = 0; p < NP; ++p) Slev += klev[p] * smLev[p];
        float Stime = 0.f;
        #pragma unroll
        for (int t = 0; t < NT; ++t) Stime += ktime[t] * smTime[t];

        const float inv = 1.f / (Sarea * Slev * Stime + 1e-4f);

        float* w = ws;
        #pragma unroll
        for (int y = 0; y < NY; ++y) w[WS_KLAT + fk * NY + y] = klat[y];
        #pragma unroll
        for (int x = 0; x < NX; ++x) w[WS_KLON + fk * NX + x] = klon[x];
        #pragma unroll
        for (int p = 0; p < NP; ++p) w[WS_KLEV + fk * NP + p] = klev[p];
        #pragma unroll
        for (int t = 0; t < NT; ++t) w[WS_KTIME + fk * NT + t] = ktime[t];
        w[WS_INV + fk] = inv;
    }
}

// ---------------------------------------------------------------------------
// Kernel 2: main contraction. grid=(B*F), block=128 (4 waves).
// ---------------------------------------------------------------------------
__global__ __launch_bounds__(128) void pkl_main_kernel(
    const float* __restrict__ field,   // (B,F,Y,X,P,T) -> (B*F, 256, 128)
    const float* __restrict__ darea,   // (B,Y,X)
    const float* __restrict__ dlev,    // (B,P)
    const float* __restrict__ dtime,   // (B,T)
    const float* __restrict__ ws,
    float* __restrict__ out)           // (B, F*K)
{
    __shared__ float sC[NK * NPT];     // C[k][pt] combined lev/time kernel * quad (8 KB)
    __shared__ float sDarea[NYX];
    __shared__ float sKlat[NK * NY];   // [k][y] for this f
    __shared__ float sKlon[NK * NX];   // [k][x]
    __shared__ float sFeats[NK];

    const int bf  = blockIdx.x;        // b*F + f
    const int b   = bf >> 2;           // F = 4
    const int f   = bf & 3;
    const int tid = threadIdx.x;

    // ---- fill LDS ----------------------------------------------------------
    sDarea[tid]       = darea[b * NYX + tid];
    sDarea[tid + 128] = darea[b * NYX + tid + 128];
    sKlat[tid]        = ws[WS_KLAT + f * (NK * NY) + tid];
    sKlat[tid + 128]  = ws[WS_KLAT + f * (NK * NY) + tid + 128];
    sKlon[tid]        = ws[WS_KLON + f * (NK * NX) + tid];
    sKlon[tid + 128]  = ws[WS_KLON + f * (NK * NX) + tid + 128];
    if (tid < NK) sFeats[tid] = 0.f;

    // C[k][pt] = klev[f,k,p]*dlev[b,p] * ktime[f,k,t]*dtime[b,t]  (2048 entries)
    #pragma unroll
    for (int j = 0; j < 16; ++j) {
        const int i  = tid * 16 + j;
        const int k  = i >> 7;
        const int pt = i & 127;
        const int p  = pt >> 3;
        const int t  = pt & 7;
        sC[i] = ws[WS_KLEV + f * (NK * NP) + k * NP + p] * dlev[b * NP + p]
              * ws[WS_KTIME + f * (NK * NT) + k * NT + t] * dtime[b * NT + t];
    }
    __syncthreads();

    // ---- per-wave WMMA GEMM ------------------------------------------------
    const int wave = tid >> 5;
    const int lane = tid & 31;
    const int n    = lane & 15;   // output k index (GEMM column)
    const int hi   = lane >> 4;   // lane half

    // Build the 4 B-fragments (32x16 f16, shared across this wave's M-tiles).
    // k order used here (contiguous, same as A): lane%16 = column N; VGPR v holds K = 2v+16*hi.
    v16h bfrag[4];
    #pragma unroll
    for (int pb = 0; pb < 4; ++pb) {
        #pragma unroll
        for (int v = 0; v < 8; ++v) {
            const int k0 = pb * 32 + 16 * hi + 2 * v;
            bfrag[pb][2 * v]     = (_Float16)sC[n * NPT + k0];
            bfrag[pb][2 * v + 1] = (_Float16)sC[n * NPT + k0 + 1];
        }
    }

    const float* fbase = field + (size_t)bf * NYX * NPT;
    float partial = 0.f;

    // 16 M-tiles of 16 yx-rows; 4 per wave. EXEC is all-ones throughout.
    for (int mt = wave; mt < 16; mt += 4) {
        const float* frow = fbase + (size_t)(mt * 16 + (lane & 15)) * NPT;

        v8f acc = {};
        #pragma unroll
        for (int pb = 0; pb < 4; ++pb) {
            // A fragment (16x32 f16): lane%16 = row M;
            // VGPR v(0..7): K = 2v+16*hi
            v16h a;
            const float* fp0 = frow + pb * 32 + 16 * hi;
            const float* fp1 = frow + pb * 32 + 8 + 16 * hi;
            #pragma unroll
            for (int j = 0; j < 8; ++j) a[j]     = (_Float16)fp0[j];
            #pragma unroll
            for (int j = 0; j < 8; ++j) a[8 + j] = (_Float16)fp1[j];

            acc = __builtin_amdgcn_wmma_f32_16x16x32_f16(
                false, a, false, bfrag[pb], (short)0, acc, false, false);
        }

        // Stage B: fold (y,x) with klat*klon*darea in f32.
        // C/D layout: lane holds column N=n, VGPR v -> row M = v + 8*hi.
        #pragma unroll
        for (int v = 0; v < 8; ++v) {
            const int yx = mt * 16 + v + 8 * hi;
            partial += acc[v] * sKlat[n * NY + (yx >> 4)]
                              * sKlon[n * NX + (yx & 15)]
                              * sDarea[yx];
        }
    }

    atomicAdd(&sFeats[n], partial);   // ds_add_f32: 8 partials per k
    __syncthreads();

    if (tid < NK)
        out[bf * NK + tid] = sFeats[tid] * ws[WS_INV + f * NK + tid];
}

// ---------------------------------------------------------------------------
extern "C" void kernel_launch(void* const* d_in, const int* in_sizes, int n_in,
                              void* d_out, int out_size, void* d_ws, size_t ws_size,
                              hipStream_t stream) {
    const float* fieldpatch = (const float*)d_in[0];
    const float* dareapatch = (const float*)d_in[1];
    const float* dlevpatch  = (const float*)d_in[2];
    const float* dtimepatch = (const float*)d_in[3];
    const float* lat_mean   = (const float*)d_in[4];
    const float* lat_logstd = (const float*)d_in[5];
    const float* lon_mean   = (const float*)d_in[6];
    const float* lon_logstd = (const float*)d_in[7];
    const float* lev_mean   = (const float*)d_in[8];
    const float* lev_logstd = (const float*)d_in[9];
    const float* time_ltau  = (const float*)d_in[10];

    float* ws  = (float*)d_ws;
    float* out = (float*)d_out;

    pkl_prep_kernel<<<1, 256, 0, stream>>>(
        dareapatch, dlevpatch, dtimepatch,
        lat_mean, lat_logstd, lon_mean, lon_logstd,
        lev_mean, lev_logstd, time_ltau, ws);

    pkl_main_kernel<<<NB * NF, 128, 0, stream>>>(
        fieldpatch, dareapatch, dlevpatch, dtimepatch, ws, out);
}
